// DeepseekV2MoE_29600914604509
// MI455X (gfx1250) — compile-verified, matrix operand order checked by reading, not run on hardware
//
#include <hip/hip_runtime.h>

// ---------------------------------------------------------------------------
// DeepSeek-V2 MoE layer for gfx1250 (MI455X): wave32 + V_WMMA_F32_16X16X32_BF16
// K-pipelined, LDS double-buffered grouped GEMMs (K-loop unrolled x2 so the
// buffer parity is compile-time const); async-to-LDS staging in gemm2.
// T=1024 D=2048 E=64 I=1408 K=6 G=8 TG=3 SHI=2816 RSF=2.5
// NOTE: all K dims here (2048, 2816, 1408) have an even number of 32-steps.
// ---------------------------------------------------------------------------

typedef __attribute__((ext_vector_type(16))) __bf16 v16bf;
typedef __attribute__((ext_vector_type(8)))  __bf16 v8bf;
typedef __attribute__((ext_vector_type(8)))  float  v8f;

#define NT 1024
#define ND 2048
#define NE 64
#define NI 1408
#define NK 6
#define NSHI 2816
#define RSF_C 2.5f

#define LDS_ROW 48          // bf16 elements per LDS row (96 B, 16B-aligned chunks)
#define LDS_BUF (128 * LDS_ROW)

__device__ inline v8bf cvt8(float4 a, float4 b) {
    v8bf r;
    r[0] = (__bf16)a.x; r[1] = (__bf16)a.y; r[2] = (__bf16)a.z; r[3] = (__bf16)a.w;
    r[4] = (__bf16)b.x; r[5] = (__bf16)b.y; r[6] = (__bf16)b.z; r[7] = (__bf16)b.w;
    return r;
}
__device__ inline v8bf zero8() {
    v8bf r;
    #pragma unroll
    for (int i = 0; i < 8; i++) r[i] = (__bf16)0.0f;
    return r;
}
#define CAT16(lo, hi) __builtin_shufflevector(lo, hi, 0,1,2,3,4,5,6,7,8,9,10,11,12,13,14,15)

// ---------------------------------------------------------------------------
// Router logits: logits[T,E] = x @ gate_w^T   (tiny GEMM, LDS-tiled VALU)
// ---------------------------------------------------------------------------
__global__ __launch_bounds__(256)
void moe_router_logits(const float* __restrict__ x, const float* __restrict__ gw,
                       float* __restrict__ logits) {
    __shared__ float xs[16][64];
    const int t0 = blockIdx.x * 16;
    const int tid = threadIdx.x;
    const int e = tid & 63, tg = tid >> 6;
    float acc[4] = {0.f, 0.f, 0.f, 0.f};
    for (int k0 = 0; k0 < ND; k0 += 64) {
        __syncthreads();
        const int r = tid >> 4, c4 = tid & 15;
        *(float4*)&xs[r][c4 * 4] = *(const float4*)&x[(size_t)(t0 + r) * ND + k0 + c4 * 4];
        __syncthreads();
        for (int kk = 0; kk < 64; kk++) {
            const float w = gw[(size_t)e * ND + k0 + kk];
            #pragma unroll
            for (int j = 0; j < 4; j++) acc[j] += xs[tg * 4 + j][kk] * w;
        }
    }
    #pragma unroll
    for (int j = 0; j < 4; j++) logits[(size_t)(t0 + tg * 4 + j) * NE + e] = acc[j];
}

__global__ void moe_zero_ints(int* p, int n) {
    int i = blockIdx.x * blockDim.x + threadIdx.x;
    if (i < n) p[i] = 0;
}

// ---------------------------------------------------------------------------
// Grouped top-k + per-expert bucket build (1 thread / token).
// ---------------------------------------------------------------------------
__global__ void moe_grouped_topk(const float* __restrict__ logits, const float* __restrict__ bias,
                                 int* __restrict__ cnt, int* __restrict__ bTok,
                                 float* __restrict__ bW, int* __restrict__ bSlot) {
    const int t = blockIdx.x * blockDim.x + threadIdx.x;
    if (t >= NT) return;
    float sc[NE], sfc[NE];
    for (int e = 0; e < NE; e++) {
        const float l = logits[(size_t)t * NE + e];
        const float s = 1.0f / (1.0f + __expf(-l));
        sc[e] = s; sfc[e] = s + bias[e];
    }
    float gs[8];
    for (int g = 0; g < 8; g++) {
        float m1 = -1e30f, m2 = -1e30f;
        for (int i = 0; i < 8; i++) {
            const float v = sfc[g * 8 + i];
            if (v > m1) { m2 = m1; m1 = v; } else if (v > m2) { m2 = v; }
        }
        gs[g] = m1 + m2;
    }
    unsigned gm = 0;
    for (int it = 0; it < 3; it++) {
        float best = -1e30f; int bi = 0;
        for (int g = 0; g < 8; g++)
            if (!((gm >> g) & 1u) && gs[g] > best) { best = gs[g]; bi = g; }
        gm |= 1u << bi;
    }
    float wsum = 0.f; int ids[NK]; float wts[NK];
    unsigned long long taken = 0ull;
    for (int it = 0; it < NK; it++) {
        float best = -1e30f; int bi = 0;
        for (int e = 0; e < NE; e++)
            if (((gm >> (e >> 3)) & 1u) && !((taken >> e) & 1ull) && sfc[e] > best) {
                best = sfc[e]; bi = e;
            }
        taken |= 1ull << bi;
        ids[it] = bi; wts[it] = sc[bi]; wsum += sc[bi];
    }
    const float inv = 1.0f / (wsum + 1e-20f);
    for (int k = 0; k < NK; k++) {
        const int e = ids[k];
        const int pos = atomicAdd(&cnt[e], 1);
        bTok[e * NT + pos] = t;
        bW[e * NT + pos] = wts[k] * inv;
        bSlot[e * NT + pos] = t * NK + k;
    }
}

// ---------------------------------------------------------------------------
// WMMA GEMM #1: C_bf16 = A_f32(gathered) @ B_f32[N,K]^T
// 128x128 block tile, 8 waves, wave = 8 M-tiles x 16 cols.
// Double-buffered LDS A tile, K-loop unrolled x2 (const buffer parity);
// B fragment and A staging pipelined one K-step ahead of the WMMAs.
// ---------------------------------------------------------------------------
__global__ __launch_bounds__(256)
void moe_wmma_gemm_af32(const float* __restrict__ A, const float* __restrict__ Bw,
                        __bf16* __restrict__ C,
                        const int* __restrict__ aIdx, const int* __restrict__ cIdx,
                        const int* __restrict__ counts, int rowsConst,
                        int Kdim, int Ntot, int cap, int numMB, long long bStride) {
    __shared__ __attribute__((aligned(16))) __bf16 As[2 * LDS_BUF];
    const int e = blockIdx.x / numMB;
    const int mb = blockIdx.x % numMB;
    const int rows = counts ? counts[e] : rowsConst;
    const int m0 = mb * 128;
    if (m0 >= rows) return;
    const int tid = threadIdx.x, lane = tid & 31, wave = tid >> 5;
    const int n0 = blockIdx.y * 128 + wave * 16;
    const float* B = Bw + (long long)e * bStride + (long long)n0 * Kdim;

    const int r = tid >> 1, half = tid & 1;          // A staging: 2 thr/row, 16 f32 each
    const int mrow = m0 + r;
    const float* arow = nullptr;
    if (mrow < rows) {
        const int ar = aIdx ? aIdx[e * cap + mrow] : mrow;
        arow = A + (long long)ar * Kdim;
    }
    const int nLoc = lane & 15;
    const int kSel = (lane < 16) ? 0 : 8;            // 16-bit A/B lane K-split
    const int arL = lane & 15;
    const float* bRow = B + (long long)nLoc * Kdim;
    __bf16* myDst0 = As + r * LDS_ROW + half * 16;

    // zero invalid rows of both buffers once
    if (!arow) {
        const v8bf z = zero8();
        *(v8bf*)(myDst0) = z;             *(v8bf*)(myDst0 + 8) = z;
        *(v8bf*)(myDst0 + LDS_BUF) = z;   *(v8bf*)(myDst0 + LDS_BUF + 8) = z;
    } else {                               // stage k0 = 0 into buffer 0
        const float4* p = (const float4*)(arow + half * 16);
        *(v8bf*)(myDst0) = cvt8(p[0], p[1]);
        *(v8bf*)(myDst0 + 8) = cvt8(p[2], p[3]);
    }
    float4 br0, br1, br2, br3;                        // B raw pipeline (k0 = 0)
    {
        const float4* p0 = (const float4*)(bRow + kSel);
        const float4* p1 = (const float4*)(bRow + kSel + 16);
        br0 = p0[0]; br1 = p0[1]; br2 = p1[0]; br3 = p1[1];
    }
    __syncthreads();

    v8f acc[8];
    #pragma unroll
    for (int i = 0; i < 8; i++)
        #pragma unroll
        for (int v = 0; v < 8; v++) acc[i][v] = 0.f;

    // One pipeline step: compute from CURBUF at K-offset, stage K0+32 -> NXTBUF.
    #define GEMM1_STEP(CURBUF, NXTBUF, K0, DO_STAGE)                              \
    {                                                                             \
        const __bf16* Ab = As + (CURBUF) * LDS_BUF;                               \
        v16bf afrag[8];                                                           \
        _Pragma("unroll")                                                         \
        for (int mt = 0; mt < 8; mt++) {                                          \
            const __bf16* ap = Ab + (mt * 16 + arL) * LDS_ROW + kSel;             \
            afrag[mt] = CAT16(*(const v8bf*)ap, *(const v8bf*)(ap + 16));         \
        }                                                                         \
        const v16bf bfrag = CAT16(cvt8(br0, br1), cvt8(br2, br3));                \
        float4 ar0, ar1, ar2, ar3;                                                \
        if (DO_STAGE) {                                                           \
            const float4* p0 = (const float4*)(bRow + (K0) + 32 + kSel);          \
            const float4* p1 = (const float4*)(bRow + (K0) + 32 + kSel + 16);     \
            br0 = p0[0]; br1 = p0[1]; br2 = p1[0]; br3 = p1[1];                   \
            if (arow) {                                                           \
                const float4* p = (const float4*)(arow + (K0) + 32 + half * 16);  \
                ar0 = p[0]; ar1 = p[1]; ar2 = p[2]; ar3 = p[3];                   \
            }                                                                     \
        }                                                                         \
        _Pragma("unroll")                                                         \
        for (int mt = 0; mt < 8; mt++)                                            \
            acc[mt] = __builtin_amdgcn_wmma_f32_16x16x32_bf16(                    \
                false, afrag[mt], false, bfrag, (short)0, acc[mt], false, false); \
        if ((DO_STAGE) && arow) {                                                 \
            __bf16* dst = myDst0 + (NXTBUF) * LDS_BUF;                            \
            *(v8bf*)dst = cvt8(ar0, ar1);                                         \
            *(v8bf*)(dst + 8) = cvt8(ar2, ar3);                                   \
        }                                                                         \
        __syncthreads();                                                          \
    }

    for (int k0 = 0; k0 < Kdim; k0 += 64) {           // K/32 is even for all GEMMs
        GEMM1_STEP(0, 1, k0, true);
        GEMM1_STEP(1, 0, k0 + 32, (k0 + 64 < Kdim));
    }
    #undef GEMM1_STEP

    // D layout: lane<16 -> M=v, lane>=16 -> M=8+v; N = lane&15
    const int mOff = (lane < 16) ? 0 : 8;
    const int nCol = n0 + (lane & 15);
    #pragma unroll
    for (int mt = 0; mt < 8; mt++)
        #pragma unroll
        for (int v = 0; v < 8; v++) {
            const int mg = m0 + mt * 16 + mOff + v;
            if (mg < rows) {
                const int cr = cIdx ? cIdx[e * cap + mg] : mg;
                C[(long long)cr * Ntot + nCol] = (__bf16)acc[mt][v];
            }
        }
}

// ---------------------------------------------------------------------------
// WMMA GEMM #2: Out_f32 = A_bf16(gathered) @ B_f32[N,K]^T
// A staging uses gfx1250 async-to-LDS (ASYNCcnt), no VGPR round trip.
// rowScale != null -> atomicAdd(scaleMul * rowScale[m] * val) scattered rows;
// rowScale == null -> plain store (initializes d_out with the shared MLP).
// ---------------------------------------------------------------------------
__device__ inline void async_cp16(unsigned ldsOff, const void* gaddr) {
    asm volatile("global_load_async_to_lds_b128 %0, %1, off"
                 :: "v"(ldsOff), "v"(gaddr) : "memory");
}
__device__ inline void wait_async0() {
    asm volatile("s_wait_asynccnt 0x0" ::: "memory");
}

__global__ __launch_bounds__(256)
void moe_wmma_gemm_abf16(const __bf16* __restrict__ A, const float* __restrict__ Bw,
                         float* __restrict__ Out,
                         const int* __restrict__ aIdx, const int* __restrict__ oIdx,
                         const float* __restrict__ rowScale,
                         const int* __restrict__ counts, int rowsConst,
                         int Kdim, int Ntot, int cap, int numMB, long long bStride,
                         float scaleMul) {
    __shared__ __attribute__((aligned(16))) __bf16 As[2 * LDS_BUF];
    const int e = blockIdx.x / numMB;
    const int mb = blockIdx.x % numMB;
    const int rows = counts ? counts[e] : rowsConst;
    const int m0 = mb * 128;
    if (m0 >= rows) return;
    const int tid = threadIdx.x, lane = tid & 31, wave = tid >> 5;
    const int n0 = blockIdx.y * 128 + wave * 16;
    const float* B = Bw + (long long)e * bStride + (long long)n0 * Kdim;

    const int r = tid >> 1, half = tid & 1;           // 2 thr/row, 16 bf16 each
    const int mrow = m0 + r;
    const __bf16* arow = nullptr;
    if (mrow < rows) {
        const int ar = aIdx ? aIdx[e * cap + mrow] : mrow;
        arow = A + (long long)ar * Kdim;
    }
    const int nLoc = lane & 15;
    const int kSel = (lane < 16) ? 0 : 8;
    const int arL = lane & 15;
    const float* bRow = B + (long long)nLoc * Kdim;
    __bf16* myDst0 = As + r * LDS_ROW + half * 16;
    const unsigned myLds0 = (unsigned)(unsigned long long)(const void*)myDst0;

    if (!arow) {
        const v8bf z = zero8();
        *(v8bf*)(myDst0) = z;             *(v8bf*)(myDst0 + 8) = z;
        *(v8bf*)(myDst0 + LDS_BUF) = z;   *(v8bf*)(myDst0 + LDS_BUF + 8) = z;
    } else {                               // async-stage k0 = 0 into buffer 0
        async_cp16(myLds0, arow + half * 16);
        async_cp16(myLds0 + 16u, arow + half * 16 + 8);
    }
    float4 br0, br1, br2, br3;
    {
        const float4* p0 = (const float4*)(bRow + kSel);
        const float4* p1 = (const float4*)(bRow + kSel + 16);
        br0 = p0[0]; br1 = p0[1]; br2 = p1[0]; br3 = p1[1];
    }
    wait_async0();
    __syncthreads();

    v8f acc[8];
    #pragma unroll
    for (int i = 0; i < 8; i++)
        #pragma unroll
        for (int v = 0; v < 8; v++) acc[i][v] = 0.f;

    #define GEMM2_STEP(CURBUF, NXTBUF, K0, DO_STAGE)                              \
    {                                                                             \
        const __bf16* Ab = As + (CURBUF) * LDS_BUF;                               \
        v16bf afrag[8];                                                           \
        _Pragma("unroll")                                                         \
        for (int mt = 0; mt < 8; mt++) {                                          \
            const __bf16* ap = Ab + (mt * 16 + arL) * LDS_ROW + kSel;             \
            afrag[mt] = CAT16(*(const v8bf*)ap, *(const v8bf*)(ap + 16));         \
        }                                                                         \
        const v16bf bfrag = CAT16(cvt8(br0, br1), cvt8(br2, br3));                \
        if (DO_STAGE) {                                                           \
            const float4* p0 = (const float4*)(bRow + (K0) + 32 + kSel);          \
            const float4* p1 = (const float4*)(bRow + (K0) + 32 + kSel + 16);     \
            br0 = p0[0]; br1 = p0[1]; br2 = p1[0]; br3 = p1[1];                   \
            if (arow) {                                                           \
                const unsigned dst = myLds0 + (unsigned)((NXTBUF) * LDS_BUF * 2); \
                const __bf16* g = arow + (K0) + 32 + half * 16;                   \
                async_cp16(dst, g);                                               \
                async_cp16(dst + 16u, g + 8);                                     \
            }                                                                     \
        }                                                                         \
        _Pragma("unroll")                                                         \
        for (int mt = 0; mt < 8; mt++)                                            \
            acc[mt] = __builtin_amdgcn_wmma_f32_16x16x32_bf16(                    \
                false, afrag[mt], false, bfrag, (short)0, acc[mt], false, false); \
        if (DO_STAGE) wait_async0();                                              \
        __syncthreads();                                                          \
    }

    for (int k0 = 0; k0 < Kdim; k0 += 64) {           // K/32 is even for all GEMMs
        GEMM2_STEP(0, 1, k0, true);
        GEMM2_STEP(1, 0, k0 + 32, (k0 + 64 < Kdim));
    }
    #undef GEMM2_STEP

    const int mOff = (lane < 16) ? 0 : 8;
    const int nCol = n0 + (lane & 15);
    #pragma unroll
    for (int mt = 0; mt < 8; mt++)
        #pragma unroll
        for (int v = 0; v < 8; v++) {
            const int mg = m0 + mt * 16 + mOff + v;
            if (mg < rows) {
                const float val = acc[mt][v];
                if (rowScale) {
                    const int t = oIdx[e * cap + mg];
                    const float s = scaleMul * rowScale[e * cap + mg];
                    atomicAdd(&Out[(long long)t * Ntot + nCol], s * val);
                } else {
                    Out[(long long)mg * Ntot + nCol] = val;
                }
            }
        }
}

// ---------------------------------------------------------------------------
// SiLU(gate) * up, bf16 -> bf16.  gu: [R, 2N], h: [R, N]
// ---------------------------------------------------------------------------
__global__ void moe_silu_mul(const __bf16* __restrict__ gu, __bf16* __restrict__ h,
                             int R, int N) {
    const int idx = blockIdx.x * blockDim.x + threadIdx.x;
    if (idx >= R * N) return;
    const int rr = idx / N, c = idx % N;
    const float g = (float)gu[(long long)rr * 2 * N + c];
    const float u = (float)gu[(long long)rr * 2 * N + N + c];
    const float s = g / (1.0f + __expf(-g));
    h[idx] = (__bf16)(s * u);
}

// ---------------------------------------------------------------------------
extern "C" void kernel_launch(void* const* d_in, const int* in_sizes, int n_in,
                              void* d_out, int out_size, void* d_ws, size_t ws_size,
                              hipStream_t stream) {
    const float* x    = (const float*)d_in[0];   // [T, D]
    const float* gw   = (const float*)d_in[1];   // [E, D]
    const float* be   = (const float*)d_in[2];   // [E]
    const float* w13  = (const float*)d_in[3];   // [E, 2I, D]
    const float* w2   = (const float*)d_in[4];   // [E, D, I]
    const float* sw13 = (const float*)d_in[5];   // [2*SHI, D]
    const float* sw2  = (const float*)d_in[6];   // [D, SHI]
    float* out = (float*)d_out;                  // [T, D]

    char* ws = (char*)d_ws;
    size_t off = 0;
    auto alloc = [&](size_t bytes) -> void* {
        void* p = ws + off;
        off = (off + bytes + 255) & ~(size_t)255;
        return p;
    };
    float* logits = (float*)alloc((size_t)NT * NE * 4);
    int*   cnt    = (int*)  alloc((size_t)NE * 4);
    int*   bTok   = (int*)  alloc((size_t)NE * NT * 4);
    float* bW     = (float*)alloc((size_t)NE * NT * 4);
    int*   bSlot  = (int*)  alloc((size_t)NE * NT * 4);
    __bf16* guR   = (__bf16*)alloc((size_t)NT * NK * 2 * NI * 2);   // [6144, 2816] bf16
    __bf16* hR    = (__bf16*)alloc((size_t)NT * NK * NI * 2);       // [6144, 1408] bf16
    __bf16* guS   = (__bf16*)alloc((size_t)NT * 2 * NSHI * 2);      // [1024, 5632] bf16
    __bf16* hS    = (__bf16*)alloc((size_t)NT * NSHI * 2);          // [1024, 2816] bf16

    // 1) routing
    moe_router_logits<<<NT / 16, 256, 0, stream>>>(x, gw, logits);
    moe_zero_ints<<<1, NE, 0, stream>>>(cnt, NE);
    moe_grouped_topk<<<NT / 128, 128, 0, stream>>>(logits, be, cnt, bTok, bW, bSlot);

    // 2) shared experts MLP (initializes d_out with plain stores)
    moe_wmma_gemm_af32<<<dim3(8, (2 * NSHI) / 128), 256, 0, stream>>>(
        x, sw13, guS, nullptr, nullptr, nullptr, NT, ND, 2 * NSHI, 0, 8, 0LL);
    moe_silu_mul<<<(NT * NSHI + 255) / 256, 256, 0, stream>>>(guS, hS, NT, NSHI);
    moe_wmma_gemm_abf16<<<dim3(8, ND / 128), 256, 0, stream>>>(
        hS, sw2, out, nullptr, nullptr, nullptr, nullptr, NT, NSHI, ND, 0, 8, 0LL, 1.0f);

    // 3) routed experts: grouped GEMMs over per-expert token buckets
    moe_wmma_gemm_af32<<<dim3(NE * 8, (2 * NI) / 128), 256, 0, stream>>>(
        x, w13, guR, bTok, bSlot, cnt, 0, ND, 2 * NI, NT, 8, (long long)(2 * NI) * ND);
    moe_silu_mul<<<(NT * NK * NI + 255) / 256, 256, 0, stream>>>(guR, hR, NT * NK, NI);
    moe_wmma_gemm_abf16<<<dim3(NE * 8, ND / 128), 256, 0, stream>>>(
        hR, w2, out, bSlot, bTok, bW, cnt, 0, NI, ND, NT, 8, (long long)ND * NI, RSF_C);
}
